// DKTtempoModel_with_x_83545703842332
// MI455X (gfx1250) — compile-verified, matrix operand order checked by reading, not run on hardware
//
#include <hip/hip_runtime.h>
#include <hip/hip_bf16.h>

// ---------------------------------------------------------------------------
// Problem constants (from reference)
// ---------------------------------------------------------------------------
#define BB 64
#define TT 512
#define KK 512
#define FF 1024
#define EE 256
#define HH 512
#define GG 2048          // 4*H
#define ROWS (BB*TT)     // 32768

#define NBLK_LSTM 32
#define WRT_LD 520       // 512 + 8 pad (keeps 16B alignment, staggers banks)

typedef __attribute__((ext_vector_type(16))) __bf16 v16bf;
typedef __attribute__((ext_vector_type(8)))  float  v8f;
typedef __attribute__((ext_vector_type(16))) unsigned short v16us;
typedef __attribute__((ext_vector_type(8)))  unsigned short v8us;
typedef int v4i __attribute__((vector_size(16)));   // matches async builtin pointee
typedef __attribute__((ext_vector_type(4))) unsigned int u32x4;
typedef __attribute__((ext_vector_type(8))) unsigned int u32x8;
typedef __attribute__((ext_vector_type(4))) int i32x4;
typedef __attribute__((ext_vector_type(8))) int i32x8;

// ---------------------------------------------------------------------------
// Optional CDNA5 async global->LDS path (guarded; fallback compiles anyway)
// ---------------------------------------------------------------------------
#if defined(__has_builtin)
#if __has_builtin(__builtin_amdgcn_global_load_async_to_lds_b128) && \
    __has_builtin(__builtin_amdgcn_s_wait_asynccnt)
#define HAVE_ASYNC_LDS 1
#endif
#if __has_builtin(__builtin_amdgcn_tensor_load_to_lds) && \
    __has_builtin(__builtin_amdgcn_s_wait_tensorcnt)
#define HAVE_TDM 1
#endif
#endif
#ifndef HAVE_ASYNC_LDS
#define HAVE_ASYNC_LDS 0
#endif
#ifndef HAVE_TDM
#define HAVE_TDM 0
#endif

#if HAVE_ASYNC_LDS
// Per-lane 16B global->LDS async transfer.  Generic->AS casts go through
// integers: AS1 keeps the 64-bit value; generic LDS addresses carry the
// 32-bit LDS offset in the low DWORD (flat-aperture mapping).  Builtin
// pointee type is v4i (from hipcc diagnostic).
__device__ __forceinline__ void async_copy_b128(const void* g, void* l) {
  __builtin_amdgcn_global_load_async_to_lds_b128(
      (__attribute__((address_space(1))) v4i*)(unsigned long long)g,
      (__attribute__((address_space(3))) v4i*)(unsigned)(unsigned long long)l,
      0, 0);
}
#endif

// ---------------------------------------------------------------------------
// Helpers
// ---------------------------------------------------------------------------
__device__ __forceinline__ unsigned short f32_to_bf16(float f) {
  union { float f; unsigned u; } v; v.f = f;
  unsigned u = v.u;
  unsigned r = u + 0x7FFFu + ((u >> 16) & 1u);   // round-to-nearest-even
  return (unsigned short)(r >> 16);
}

__device__ __forceinline__ float bf16_to_f32(unsigned short h) {
  union { unsigned u; float f; } v; v.u = ((unsigned)h) << 16;
  return v.f;
}

__device__ __forceinline__ float sigmoid_f(float x) {
  return 1.0f / (1.0f + __expf(-x));
}

__device__ __forceinline__ v8f zero_v8f() {
  v8f z = {0.f,0.f,0.f,0.f,0.f,0.f,0.f,0.f};
  return z;
}

// Load a 16x32 bf16 A-style fragment (row-major source, leading dim `ld`
// in elements).  Per CDNA5 ISA 7.12.2: lane<16 holds row=lane, K 0..7 in
// v[0..3] halves and K 16..23 in v[4..7]; lane>=16 holds K 8..15 / 24..31.
// Two 16-byte loads per lane.  B fragments use the same pattern from an
// N-major (transposed) staging buffer.
__device__ __forceinline__ v16us load_frag(const unsigned short* base, int ld) {
  const int lane = threadIdx.x & 31;
  const unsigned short* p = base + (size_t)(lane & 15) * ld + ((lane & 16) ? 8 : 0);
  v8us lo = *(const v8us*)(p);
  v8us hi = *(const v8us*)(p + 16);
  v16us f;
#pragma unroll
  for (int i = 0; i < 8; ++i) { f[i] = lo[i]; f[8 + i] = hi[i]; }
  return f;
}

__device__ __forceinline__ v8f wmma_bf16(v16us a, v16us b, v8f c) {
  return __builtin_amdgcn_wmma_f32_16x16x32_bf16(
      false, __builtin_bit_cast(v16bf, a),
      false, __builtin_bit_cast(v16bf, b),
      (short)0, c, false, false);
}

// Grid-wide barrier via monotone counter; all 32 blocks resident on MI455X.
__device__ __forceinline__ void grid_barrier(unsigned* bar, int nblk, unsigned& epoch) {
  __syncthreads();
  if (threadIdx.x == 0) {
    __threadfence();
    atomicAdd(bar, 1u);
    epoch += (unsigned)nblk;
    while (atomicAdd(bar, 0u) < epoch) {
      __builtin_amdgcn_s_sleep(1);
    }
  }
  __syncthreads();
  __threadfence();
}

// ---------------------------------------------------------------------------
// K0: zero output, h double-buffer, barrier counter
// ---------------------------------------------------------------------------
__global__ void k_zero(float* out, unsigned short* hbf, unsigned* bar) {
  int i = blockIdx.x * 256 + threadIdx.x;
  if (i < ROWS) out[i] = 0.f;
  if (i < 2 * BB * HH) hbf[i] = 0;
  if (i < 64) bar[i] = 0u;
}

// ---------------------------------------------------------------------------
// K1: W_c (1024x255) -> bf16 padded to 1024x256 (col 255 = 0); b_c padded
// ---------------------------------------------------------------------------
__global__ void k_wc_pad(const float* Wc, const float* bc,
                         unsigned short* WcB, float* bcp) {
  int i = blockIdx.x * 256 + threadIdx.x;
  if (i < FF * EE) {
    int r = i >> 8, c = i & 255;
    WcB[i] = (c < EE - 1) ? f32_to_bf16(Wc[r * (EE - 1) + c]) : (unsigned short)0;
  }
  if (i < EE) bcp[i] = (i < EE - 1) ? bc[i] : 0.f;
}

// ---------------------------------------------------------------------------
// K2: generic f32 -> bf16 conversion
// ---------------------------------------------------------------------------
__global__ void k_cvt_bf16(const float* src, unsigned short* dst, int n) {
  int i = blockIdx.x * 256 + threadIdx.x;
  if (i < n) dst[i] = f32_to_bf16(src[i]);
}

// ---------------------------------------------------------------------------
// K3: per-(b,t) row sum of x and nonzero mask.  One block per row.
// ---------------------------------------------------------------------------
__global__ void __launch_bounds__(256) k_rowsum(const float* x, float* a, float* mx) {
  int row = blockIdx.x;
  const float* xr = x + (size_t)row * FF;
  float s = 0.f; int nz = 0;
  for (int i = threadIdx.x; i < FF; i += 256) {
    float v = xr[i]; s += v; nz |= (v != 0.f);
  }
  __shared__ float sred[256];
  __shared__ int   zred[256];
  sred[threadIdx.x] = s; zred[threadIdx.x] = nz;
  __syncthreads();
  for (int off = 128; off; off >>= 1) {
    if (threadIdx.x < off) {
      sred[threadIdx.x] += sred[threadIdx.x + off];
      zred[threadIdx.x] |= zred[threadIdx.x + off];
    }
    __syncthreads();
  }
  if (threadIdx.x == 0) { a[row] = sred[0]; mx[row] = zred[0] ? 1.f : 0.f; }
}

// ---------------------------------------------------------------------------
// K4: counts scan over T.  Thread owns (b,f); grid = B * F/256 blocks.
// ---------------------------------------------------------------------------
__global__ void __launch_bounds__(256) k_count_scan(
    const float* x, const float* a, const float* mx, unsigned short* counts) {
  int b = blockIdx.x >> 2;
  int f = ((blockIdx.x & 3) << 8) + threadIdx.x;
  float s = 0.f;
  for (int t = 0; t < TT; ++t) {
    int row = b * TT + t;
    if (t + 1 < TT)
      __builtin_prefetch(&x[(size_t)(row + 1) * FF + f], 0, 1);
    if (mx[row] != 0.f) s = a[row] * s + x[(size_t)row * FF + f];
    counts[(size_t)row * FF + f] = f32_to_bf16(s);
  }
}

// ---------------------------------------------------------------------------
// K5: generic bf16 WMMA GEMM.  C(M,N) = A(M,K_) * B(K_,N) + bias
//   mode 0: store bf16 to outb
//   mode 1: store f32 to outf
//   mode 2: y = sigmoid(v); p = y*q; row-reduce; atomicAdd into outdot[row]
// Block tile 128(M) x 64(N); 8 waves; B panel staged transposed in LDS.
// ---------------------------------------------------------------------------
__global__ void __launch_bounds__(256) k_gemm(
    const unsigned short* __restrict__ A, const unsigned short* __restrict__ Bw,
    const float* __restrict__ bias,
    float* outf, unsigned short* outb,
    const float* __restrict__ q, float* outdot,
    int N, int K_, int mode) {
  __shared__ __align__(16) unsigned short Bt[64][40];
  const int tid = threadIdx.x;
  const int lane = tid & 31, w = tid >> 5;
  const int n0 = blockIdx.x * 64;
  const int mw = blockIdx.y * 128 + w * 16;

  v8f acc[4];
#pragma unroll
  for (int j = 0; j < 4; ++j) acc[j] = zero_v8f();

  const int kkkf = tid >> 3;            // 0..31  (B-panel fill coords)
  const int nnf  = (tid & 7) << 3;      // 0..56

  for (int kk = 0; kk < K_; kk += 32) {
    __syncthreads();
    {
      v8us bv = *(const v8us*)(Bw + (size_t)(kk + kkkf) * N + n0 + nnf);
#pragma unroll
      for (int i = 0; i < 8; ++i) Bt[nnf + i][kkkf] = bv[i];
    }
    __syncthreads();
    if (kk + 32 < K_) {
      // prefetch next B panel line and next A chunk (global_prefetch_b8)
      __builtin_prefetch(Bw + (size_t)(kk + 32 + kkkf) * N + n0 + nnf, 0, 1);
      __builtin_prefetch(A + (size_t)(mw + (lane & 15)) * K_ + kk + 32 +
                             ((lane & 16) ? 8 : 0), 0, 1);
    }
    v16us fa = load_frag(A + (size_t)mw * K_ + kk, K_);
#pragma unroll
    for (int j = 0; j < 4; ++j) {
      v16us fb = load_frag(&Bt[j * 16][0], 40);
      acc[j] = wmma_bf16(fa, fb, acc[j]);
    }
  }

  const int half = (lane >> 4) * 8;
  const int nn = lane & 15;
#pragma unroll
  for (int j = 0; j < 4; ++j) {
#pragma unroll
    for (int r = 0; r < 8; ++r) {
      int row = mw + half + r;
      int col = n0 + j * 16 + nn;
      float v = acc[j][r] + bias[col];
      if (mode == 0) {
        outb[(size_t)row * N + col] = f32_to_bf16(v);
      } else if (mode == 1) {
        outf[(size_t)row * N + col] = v;
      } else {
        float y = sigmoid_f(v);
        float p = y * q[(size_t)row * N + col];
        p += __shfl_xor(p, 8, 16);
        p += __shfl_xor(p, 4, 16);
        p += __shfl_xor(p, 2, 16);
        p += __shfl_xor(p, 1, 16);
        if (nn == 0) atomicAdd(&outdot[row], p);
      }
    }
  }
}

// ---------------------------------------------------------------------------
// K6: write delta into c[:,255] and compute mask_c (treat -0 bf16 as zero).
// ---------------------------------------------------------------------------
__global__ void k_delta_maskc(unsigned short* cB, const float* delta, float* mc) {
  int row = blockIdx.x * 256 + threadIdx.x;
  if (row < ROWS) {
    cB[(size_t)row * EE + (EE - 1)] = f32_to_bf16(delta[row]);
    int nz = 0;
    const unsigned short* cr = cB + (size_t)row * EE;
    for (int i = 0; i < EE; ++i) nz |= ((cr[i] & 0x7FFF) != 0);
    mc[row] = nz ? 1.f : 0.f;
  }
}

// ---------------------------------------------------------------------------
// K7: persistent LSTM.  32 blocks x 256 threads (8 waves).
// Block owns h columns [16*blk, 16*blk+16) -> 64 gate columns (i,f,g,o x16).
// Wr slice resident in LDS (bf16, transposed, 512x64).  h exchanged through
// a double-buffered global bf16 buffer; one grid barrier per step.
//  - h staged per step via GLOBAL_LOAD_ASYNC_TO_LDS_B128 (ASYNCcnt).
//  - zx slice fetched per step via TENSOR_LOAD_TO_LDS (TDM, TENSORcnt):
//    3D tile x=16 f32, y=4 gates (stride H), z=64 rows (stride T*G), landing
//    in LDS as [m][gate*16+jj], overlapped with the WMMA loop.
// ---------------------------------------------------------------------------
__global__ void __launch_bounds__(256) k_lstm(
    const float* __restrict__ Wr, const float* __restrict__ zx,
    const float* __restrict__ mc,
    unsigned short* hbf /* 2 x 64 x 512 */, unsigned short* hseq,
    unsigned* bar) {
  __shared__ __align__(16) unsigned short WrT[64][WRT_LD];  // [nloc][k]
#if HAVE_ASYNC_LDS
  __shared__ __align__(16) unsigned short hls[BB][WRT_LD];  // h stage [m][k]
#endif
#if HAVE_TDM
  __shared__ __align__(16) float zls[BB][64];               // TDM-landed zx tile
#define ZB(m, n) zls[m][n]
#else
  __shared__ float zbuf[BB][65];
#define ZB(m, n) zbuf[m][n]
#endif
  __shared__ float cst[BB][16];

  const int tid = threadIdx.x;
  const int lane = tid & 31, w = tid >> 5;
  const int blk = blockIdx.x;
  const int j0 = blk * 16;

  // Stage transposed Wr slice (once): WrT[nloc][k] = Wr[k][gate*512 + j0 + jj]
  for (int idx = tid; idx < 64 * HH; idx += 256) {
    int nloc = idx >> 9;         // 0..63
    int k = idx & 511;
    int gcol = (nloc >> 4) * HH + j0 + (nloc & 15);
    WrT[nloc][k] = f32_to_bf16(Wr[(size_t)k * GG + gcol]);
  }
  for (int idx = tid; idx < BB * 16; idx += 256)
    cst[idx >> 4][idx & 15] = 0.f;

  unsigned epoch = 0;
  grid_barrier(bar, NBLK_LSTM, epoch);   // h buffers zeroed by k_zero

  const int m0 = (w & 3) * 16;           // batch-tile row base (M=64 -> 4 tiles)
  const int n0 = (w >> 2) * 32;          // this wave's 2 N tiles: n0, n0+16
  const int half = (lane >> 4) * 8;
  const int nn = lane & 15;

  for (int t = 0; t < TT; ++t) {
    const unsigned short* hread = hbf + (size_t)(t & 1) * BB * HH;
    unsigned short* hwrite = hbf + (size_t)((t + 1) & 1) * BB * HH;

#if HAVE_TDM
    // Issue TDM load of this step's zx slice; overlaps with the WMMA loop.
    // D# per ISA 8.3-8.6: group0 = {count|flags, lds_addr, global_addr, type},
    // group1 = {mask/data_size, tensor_dim0/1, tile_dim0/1/2, strides},
    // group2 = {tensor_dim2, ...}.  data_size=4B; tile (16,4,64);
    // strides (H, T*G); origin zx + t*G + j0.
    if (w == 0) {
      unsigned long long ga =
          (unsigned long long)(const void*)(zx + (size_t)t * GG + j0);
      unsigned lds_off = (unsigned)(unsigned long long)(const void*)&zls[0][0];
      const unsigned td0 = GG, td1 = 4u;
      const unsigned long long s0 = HH;
      const unsigned long long s1 = (unsigned long long)TT * GG;
      u32x4 g0;
      g0[0] = 1u;                                            // count=1
      g0[1] = lds_off;                                       // lds_addr
      g0[2] = (unsigned)(ga & 0xFFFFFFFFu);                  // global_addr lo
      g0[3] = (unsigned)((ga >> 32) & 0x01FFFFFFu) | (2u << 30);  // hi | type=2
      u32x8 g1;
      g1[0] = (2u << 16);                                    // data_size=4B
      g1[1] = (td0 & 0xFFFFu) << 16;                         // tensor_dim0 lo
      g1[2] = (td0 >> 16) | ((td1 & 0xFFFFu) << 16);         // td0 hi | td1 lo
      g1[3] = (td1 >> 16) | (16u << 16);                     // td1 hi | tile_dim0
      g1[4] = 4u | (64u << 16);                              // tile_dim1|tile_dim2
      g1[5] = (unsigned)(s0 & 0xFFFFFFFFu);                  // dim0_stride lo
      g1[6] = (unsigned)(s0 >> 32) | ((unsigned)(s1 & 0xFFFFu) << 16);
      g1[7] = (unsigned)(s1 >> 16);                          // dim1_stride hi
      u32x4 g2;
      g2[0] = 64u;                                           // tensor_dim2
      g2[1] = 0u; g2[2] = 0u; g2[3] = 0u;                    // dim3/stride/tile3
      u32x4 g3; g3[0] = 0u; g3[1] = 0u; g3[2] = 0u; g3[3] = 0u;
      u32x8 g4;                                              // extra group (zeros)
      g4[0] = 0u; g4[1] = 0u; g4[2] = 0u; g4[3] = 0u;
      g4[4] = 0u; g4[5] = 0u; g4[6] = 0u; g4[7] = 0u;
      __builtin_amdgcn_tensor_load_to_lds(
          g0, __builtin_bit_cast(i32x8, g1),
          __builtin_bit_cast(i32x4, g2), __builtin_bit_cast(i32x4, g3),
          __builtin_bit_cast(i32x8, g4), 0);
    }
#endif

#if HAVE_ASYNC_LDS
    // Stage h (64x512 bf16 = 64KB) into padded LDS rows: 4096 x 16B lane
    // transfers, 16 per thread, tracked by ASYNCcnt.
#pragma unroll
    for (int i = 0; i < 16; ++i) {
      int idx = tid + i * 256;           // 0..4095
      int row = idx >> 6;                // 0..63
      int c   = (idx & 63) << 3;         // element offset, 16B chunks
      async_copy_b128(hread + (size_t)row * HH + c, &hls[row][c]);
    }
#endif
#if !HAVE_TDM
    // Prefetch this step's zx slice (64 rows x 4 gate segments x 64B).
    {
      int m = tid >> 2, g = tid & 3;
      __builtin_prefetch(&zx[((size_t)m * TT + t) * GG + g * HH + j0], 0, 1);
    }
#endif
#if HAVE_ASYNC_LDS
    __builtin_amdgcn_s_wait_asynccnt(0);
    __syncthreads();
#endif

    v8f acc0 = zero_v8f(), acc1 = zero_v8f();
    for (int kk = 0; kk < HH; kk += 32) {
#if HAVE_ASYNC_LDS
      v16us fa  = load_frag(&hls[m0][kk], WRT_LD);
#else
      v16us fa  = load_frag(hread + (size_t)m0 * HH + kk, HH);
#endif
      v16us fb0 = load_frag(&WrT[n0][kk], WRT_LD);
      v16us fb1 = load_frag(&WrT[n0 + 16][kk], WRT_LD);
      acc0 = wmma_bf16(fa, fb0, acc0);
      acc1 = wmma_bf16(fa, fb1, acc1);
    }

#if HAVE_TDM
    if (w == 0) __builtin_amdgcn_s_wait_tensorcnt(0);
    __syncthreads();   // TDM tile visible to all waves before accumulate
#endif

    // z = recurrent + zx  -> ZB[m][nloc]
#pragma unroll
    for (int r = 0; r < 8; ++r) {
      int m = m0 + half + r;
      int nl0 = n0 + nn, nl1 = n0 + 16 + nn;
#if HAVE_TDM
      ZB(m, nl0) = acc0[r] + ZB(m, nl0);   // zls holds the zx values
      ZB(m, nl1) = acc1[r] + ZB(m, nl1);
#else
      size_t zrow = ((size_t)m * TT + t) * GG;
      int g0i = (nl0 >> 4) * HH + j0 + (nl0 & 15);
      int g1i = (nl1 >> 4) * HH + j0 + (nl1 & 15);
      ZB(m, nl0) = acc0[r] + zx[zrow + g0i];
      ZB(m, nl1) = acc1[r] + zx[zrow + g1i];
#endif
    }
    __syncthreads();

    // Elementwise LSTM cell for 64 batches x 16 h-cols
    for (int idx = tid; idx < BB * 16; idx += 256) {
      int m = idx >> 4, jj = idx & 15;
      float iv = sigmoid_f(ZB(m, jj));
      float fv = sigmoid_f(ZB(m, 16 + jj));
      float gv = tanhf(ZB(m, 32 + jj));
      float ov = sigmoid_f(ZB(m, 48 + jj));
      float cn = fv * cst[m][jj] + iv * gv;
      float hn = ov * tanhf(cn);
      float hv;
      if (mc[m * TT + t] != 0.f) { cst[m][jj] = cn; hv = hn; }
      else { hv = bf16_to_f32(hread[m * HH + j0 + jj]); }
      unsigned short hb = f32_to_bf16(hv);
      hwrite[m * HH + j0 + jj] = hb;
      hseq[((size_t)m * TT + t) * HH + j0 + jj] = hb;
    }

    grid_barrier(bar, NBLK_LSTM, epoch);
  }
}

// ---------------------------------------------------------------------------
// Host-side launch
// ---------------------------------------------------------------------------
extern "C" void kernel_launch(void* const* d_in, const int* in_sizes, int n_in,
                              void* d_out, int out_size, void* d_ws, size_t ws_size,
                              hipStream_t stream) {
  (void)in_sizes; (void)n_in; (void)out_size; (void)ws_size;
  const float* x      = (const float*)d_in[0];
  const float* delta  = (const float*)d_in[1];
  const float* q      = (const float*)d_in[2];
  const float* W_c    = (const float*)d_in[3];
  const float* b_c    = (const float*)d_in[4];
  const float* Wk     = (const float*)d_in[5];
  const float* Wr     = (const float*)d_in[6];
  const float* b_lstm = (const float*)d_in[7];
  const float* W_out  = (const float*)d_in[8];
  const float* b_out  = (const float*)d_in[9];
  float* out = (float*)d_out;

  // Workspace carve-up (256B aligned).  Total ~388 MB.
  char* ws = (char*)d_ws;
  size_t off = 0;
  auto carve = [&](size_t bytes) -> char* {
    char* p = ws + off;
    off = (off + bytes + 255) & ~(size_t)255;
    return p;
  };
  float*          a_sum  = (float*)carve(ROWS * 4);
  float*          maskx  = (float*)carve(ROWS * 4);
  float*          maskc  = (float*)carve(ROWS * 4);
  float*          bcpad  = (float*)carve(EE * 4);
  unsigned*       bar    = (unsigned*)carve(256);
  unsigned short* WcB    = (unsigned short*)carve((size_t)FF * EE * 2);
  unsigned short* WkB    = (unsigned short*)carve((size_t)EE * GG * 2);
  unsigned short* WoB    = (unsigned short*)carve((size_t)HH * KK * 2);
  unsigned short* hbf    = (unsigned short*)carve((size_t)2 * BB * HH * 2);
  unsigned short* counts = (unsigned short*)carve((size_t)ROWS * FF * 2);
  unsigned short* cB     = (unsigned short*)carve((size_t)ROWS * EE * 2);
  unsigned short* hseq   = (unsigned short*)carve((size_t)ROWS * HH * 2);
  float*          zxbuf  = (float*)carve((size_t)ROWS * GG * 4);

  // 0) zero out/h/barrier
  k_zero<<<256, 256, 0, stream>>>(out, hbf, bar);
  // 1) weight conversions
  k_wc_pad<<<(FF * EE + 255) / 256, 256, 0, stream>>>(W_c, b_c, WcB, bcpad);
  k_cvt_bf16<<<(EE * GG + 255) / 256, 256, 0, stream>>>(Wk, WkB, EE * GG);
  k_cvt_bf16<<<(HH * KK + 255) / 256, 256, 0, stream>>>(W_out, WoB, HH * KK);
  // 2) row sums + x mask
  k_rowsum<<<ROWS, 256, 0, stream>>>(x, a_sum, maskx);
  // 3) counts scan over T
  k_count_scan<<<BB * (FF / 256), 256, 0, stream>>>(x, a_sum, maskx, counts);
  // 4) emb = counts @ W_c + b_c  (bf16 out into c)
  k_gemm<<<dim3(EE / 64, ROWS / 128), 256, 0, stream>>>(
      counts, WcB, bcpad, nullptr, cB, nullptr, nullptr, EE, FF, 0);
  // 5) c[:,255] = delta; mask_c
  k_delta_maskc<<<(ROWS + 255) / 256, 256, 0, stream>>>(cB, delta, maskc);
  // 6) zx = c @ Wk + b_lstm (f32)
  k_gemm<<<dim3(GG / 64, ROWS / 128), 256, 0, stream>>>(
      cB, WkB, b_lstm, zxbuf, nullptr, nullptr, nullptr, GG, EE, 1);
  // 7) persistent LSTM over T
  k_lstm<<<NBLK_LSTM, 256, 0, stream>>>(Wr, zxbuf, maskc, hbf, hseq, bar);
  // 8) out = sum(sigmoid(h_seq @ W_out + b_out) * q)
  k_gemm<<<dim3(KK / 64, ROWS / 128), 256, 0, stream>>>(
      hseq, WoB, b_out, nullptr, nullptr, q, out, KK, HH, 2);
}
